// CTransformer_66048007078346
// MI455X (gfx1250) — compile-verified, matrix operand order checked by reading, not run on hardware
//
#include <hip/hip_runtime.h>
#include <hip/hip_bf16.h>

typedef __attribute__((ext_vector_type(16))) __bf16 v16bf;
typedef __attribute__((ext_vector_type(8)))  __bf16 v8bf;
typedef __attribute__((ext_vector_type(8)))  float  v8f;

#define HW_   4096
#define C_    256
#define NPIX  16384   // B*H*W = 4*64*64

__device__ __forceinline__ unsigned short f2bf(float f) {
    unsigned int u = __builtin_bit_cast(unsigned int, f);
    u += 0x7FFFu + ((u >> 16) & 1u);            // round-to-nearest-even
    return (unsigned short)(u >> 16);
}

__device__ __forceinline__ v16bf load_tile16(const unsigned short* p) {
    v8bf lo = *reinterpret_cast<const v8bf*>(p);        // K+0..7  (per-lane)
    v8bf hi = *reinterpret_cast<const v8bf*>(p + 16);   // K+16..23
    return __builtin_shufflevector(lo, hi, 0,1,2,3,4,5,6,7,8,9,10,11,12,13,14,15);
}

// ---------------- weight fp32 -> bf16 ----------------
__global__ void k_convert_bf16(const float* __restrict__ src,
                               unsigned short* __restrict__ dst, int n) {
    int i = blockIdx.x * blockDim.x + threadIdx.x;
    if (i < n) dst[i] = f2bf(src[i]);
}

// ---------------- pad-token k/v: ln_b @ W_{k,v}^T + b_{k,v} ----------------
__global__ void k_pad_kv(const float* __restrict__ ln_b,
                         const float* __restrict__ in_w,   // [768,256]
                         const float* __restrict__ in_b,   // [768]
                         float* __restrict__ padkv) {      // [512] (k then v)
    int d = blockIdx.x * blockDim.x + threadIdx.x;         // 0..511
    if (d >= 512) return;
    const float* row = in_w + (size_t)(256 + d) * C_;
    float s = in_b[256 + d];
    for (int c = 0; c < C_; ++c) s += ln_b[c] * row[c];
    padkv[d] = s;
}

// ---------------- per-pixel LayerNorm -> bf16 [NPIX, 256] ----------------
// one wave per pixel; lane owns channels [8*lane, 8*lane+8)
__global__ void k_layernorm(const float* __restrict__ src,   // [B,C,H,W]
                            const float* __restrict__ ln_w,
                            const float* __restrict__ ln_b,
                            unsigned short* __restrict__ x2) {
    int gt   = blockIdx.x * blockDim.x + threadIdx.x;
    int p    = gt >> 5;                 // pixel id
    int lane = threadIdx.x & 31;
    if (p >= NPIX) return;
    int b   = p >> 12;
    int pix = p & (HW_ - 1);
    const float* base = src + (size_t)b * C_ * HW_ + pix;

    float v[8];
    float s = 0.f;
#pragma unroll
    for (int i = 0; i < 8; ++i) {
        int c = lane * 8 + i;
        v[i] = base[(size_t)c * HW_];
        s += v[i];
    }
#pragma unroll
    for (int off = 16; off >= 1; off >>= 1) s += __shfl_xor(s, off, 32);
    float mu = s * (1.0f / 256.0f);
    float q = 0.f;
#pragma unroll
    for (int i = 0; i < 8; ++i) { float d = v[i] - mu; q += d * d; }
#pragma unroll
    for (int off = 16; off >= 1; off >>= 1) q += __shfl_xor(q, off, 32);
    float rstd = rsqrtf(q * (1.0f / 256.0f) + 1e-5f);

    unsigned int w[4];
#pragma unroll
    for (int j = 0; j < 4; ++j) {
        int c = lane * 8 + j * 2;
        float y0 = (v[j*2+0] - mu) * rstd * ln_w[c+0] + ln_b[c+0];
        float y1 = (v[j*2+1] - mu) * rstd * ln_w[c+1] + ln_b[c+1];
        w[j] = (unsigned int)f2bf(y0) | ((unsigned int)f2bf(y1) << 16);
    }
    uint4 u; u.x = w[0]; u.y = w[1]; u.z = w[2]; u.w = w[3];
    *reinterpret_cast<uint4*>(x2 + (size_t)p * C_ + lane * 8) = u;
}

// ---------------- QKV GEMM: qkv[16384,768] = x2 @ in_w^T + in_b ----------------
// bf16 WMMA, wave tile 16M x 64N, K=256
__global__ void k_gemm_qkv(const unsigned short* __restrict__ A,   // [M,256] bf16
                           const unsigned short* __restrict__ W,   // [768,256] bf16
                           const float* __restrict__ bias,         // [768]
                           float* __restrict__ D) {                // [M,768]
    const int K = 256, N = 768;
    int wave = (blockIdx.x * blockDim.x + threadIdx.x) >> 5;
    int lane = threadIdx.x & 31;
    const int tilesN = N / 64;
    if (wave >= (NPIX / 16) * tilesN) return;
    int m0 = (wave / tilesN) * 16;
    int n0 = (wave % tilesN) * 64;
    int l = lane & 15, hi = lane >> 4;

    v8f acc[4] = { {0,0,0,0,0,0,0,0},{0,0,0,0,0,0,0,0},
                   {0,0,0,0,0,0,0,0},{0,0,0,0,0,0,0,0} };
    const unsigned short* aRow = A + (size_t)(m0 + l) * K + hi * 8;
#pragma unroll
    for (int k0 = 0; k0 < K; k0 += 32) {
        v16bf a = load_tile16(aRow + k0);
#pragma unroll
        for (int t = 0; t < 4; ++t) {
            const unsigned short* bRow = W + (size_t)(n0 + t*16 + l) * K + hi * 8 + k0;
            v16bf bm = load_tile16(bRow);
            acc[t] = __builtin_amdgcn_wmma_f32_16x16x32_bf16(
                false, a, false, bm, (short)0, acc[t], false, false);
        }
    }
#pragma unroll
    for (int t = 0; t < 4; ++t) {
        int n  = n0 + t * 16 + l;
        float bb = bias[n];
        int mr = m0 + hi * 8;
#pragma unroll
        for (int r = 0; r < 8; ++r)
            D[(size_t)(mr + r) * N + n] = acc[t][r] + bb;
    }
}

// ---------------- per-pixel 9-key attention ----------------
// one wave per pixel; lane owns channels [8*lane,8*lane+8); head = lane>>2
__global__ void k_attention(const float* __restrict__ qkv,    // [NPIX,768] q|k|v
                            const float* __restrict__ padkv,  // [512] k|v
                            unsigned short* __restrict__ obf) // [NPIX,256] bf16
{
    int gt   = blockIdx.x * blockDim.x + threadIdx.x;
    int p    = gt >> 5;
    int lane = threadIdx.x & 31;
    if (p >= NPIX) return;
    int b = p >> 12, pix = p & (HW_ - 1);
    int y = pix >> 6, x = pix & 63;
    const float scale = 0.17677669529663689f;   // 32^-0.5

    const float* qp = qkv + (size_t)p * 768 + lane * 8;
    float4 q0 = *reinterpret_cast<const float4*>(qp);
    float4 q1 = *reinterpret_cast<const float4*>(qp + 4);
    float q[8] = { q0.x*scale, q0.y*scale, q0.z*scale, q0.w*scale,
                   q1.x*scale, q1.y*scale, q1.z*scale, q1.w*scale };

    const float* kptr[9];
    const float* vptr[9];
#pragma unroll
    for (int t = 0; t < 9; ++t) {
        int ny = y + t / 3 - 1, nx = x + t % 3 - 1;
        if ((unsigned)ny < 64u && (unsigned)nx < 64u) {
            size_t np = (size_t)((b << 12) + (ny << 6) + nx) * 768;
            kptr[t] = qkv + np + 256 + lane * 8;
            vptr[t] = qkv + np + 512 + lane * 8;
        } else {
            kptr[t] = padkv + lane * 8;
            vptr[t] = padkv + 256 + lane * 8;
        }
    }

    float sc[9];
#pragma unroll
    for (int t = 0; t < 9; ++t) {
        float4 k0 = *reinterpret_cast<const float4*>(kptr[t]);
        float4 k1 = *reinterpret_cast<const float4*>(kptr[t] + 4);
        float s = q[0]*k0.x + q[1]*k0.y + q[2]*k0.z + q[3]*k0.w
                + q[4]*k1.x + q[5]*k1.y + q[6]*k1.z + q[7]*k1.w;
        s += __shfl_xor(s, 1, 32);      // reduce within the 4-lane head group
        s += __shfl_xor(s, 2, 32);
        sc[t] = s;
    }
    float m = sc[0];
#pragma unroll
    for (int t = 1; t < 9; ++t) m = fmaxf(m, sc[t]);
    float wsum = 0.f;
#pragma unroll
    for (int t = 0; t < 9; ++t) { sc[t] = __expf(sc[t] - m); wsum += sc[t]; }
    float inv = 1.0f / wsum;

    float acc[8] = {0,0,0,0,0,0,0,0};
#pragma unroll
    for (int t = 0; t < 9; ++t) {
        float4 v0 = *reinterpret_cast<const float4*>(vptr[t]);
        float4 v1 = *reinterpret_cast<const float4*>(vptr[t] + 4);
        float wt = sc[t];
        acc[0] += wt * v0.x; acc[1] += wt * v0.y; acc[2] += wt * v0.z; acc[3] += wt * v0.w;
        acc[4] += wt * v1.x; acc[5] += wt * v1.y; acc[6] += wt * v1.z; acc[7] += wt * v1.w;
    }
    unsigned int w[4];
#pragma unroll
    for (int j = 0; j < 4; ++j)
        w[j] = (unsigned int)f2bf(acc[j*2] * inv)
             | ((unsigned int)f2bf(acc[j*2+1] * inv) << 16);
    uint4 u; u.x = w[0]; u.y = w[1]; u.z = w[2]; u.w = w[3];
    *reinterpret_cast<uint4*>(obf + (size_t)p * C_ + lane * 8) = u;
}

// ---------------- out-proj GEMM + bias + residual -> NCHW output ----------------
__global__ void k_gemm_out(const unsigned short* __restrict__ A,   // [M,256] bf16 (attn out)
                           const unsigned short* __restrict__ W,   // [256,256] bf16
                           const float* __restrict__ bias,         // [256]
                           const float* __restrict__ src,          // [B,C,H,W] residual
                           float* __restrict__ out) {              // [B,C,H,W]
    const int K = 256, N = 256;
    int wave = (blockIdx.x * blockDim.x + threadIdx.x) >> 5;
    int lane = threadIdx.x & 31;
    const int tilesN = N / 64;
    if (wave >= (NPIX / 16) * tilesN) return;
    int m0 = (wave / tilesN) * 16;
    int n0 = (wave % tilesN) * 64;
    int l = lane & 15, hi = lane >> 4;

    v8f acc[4] = { {0,0,0,0,0,0,0,0},{0,0,0,0,0,0,0,0},
                   {0,0,0,0,0,0,0,0},{0,0,0,0,0,0,0,0} };
    const unsigned short* aRow = A + (size_t)(m0 + l) * K + hi * 8;
#pragma unroll
    for (int k0 = 0; k0 < K; k0 += 32) {
        v16bf a = load_tile16(aRow + k0);
#pragma unroll
        for (int t = 0; t < 4; ++t) {
            const unsigned short* bRow = W + (size_t)(n0 + t*16 + l) * K + hi * 8 + k0;
            v16bf bm = load_tile16(bRow);
            acc[t] = __builtin_amdgcn_wmma_f32_16x16x32_bf16(
                false, a, false, bm, (short)0, acc[t], false, false);
        }
    }
    // lane holds 8 consecutive pixels (M) at fixed channel n -> contiguous in NCHW
    int bIdx = m0 >> 12;
    int pix  = (m0 & (HW_ - 1)) + hi * 8;
#pragma unroll
    for (int t = 0; t < 4; ++t) {
        int n  = n0 + t * 16 + l;
        float bb = bias[n];
        size_t off = (size_t)(bIdx * C_ + n) * HW_ + pix;
        const float* sp = src + off;
        float* op = out + off;
        float4 r0, r1;
        r0.x = acc[t][0] + bb + sp[0];
        r0.y = acc[t][1] + bb + sp[1];
        r0.z = acc[t][2] + bb + sp[2];
        r0.w = acc[t][3] + bb + sp[3];
        r1.x = acc[t][4] + bb + sp[4];
        r1.y = acc[t][5] + bb + sp[5];
        r1.z = acc[t][6] + bb + sp[6];
        r1.w = acc[t][7] + bb + sp[7];
        *reinterpret_cast<float4*>(op)     = r0;
        *reinterpret_cast<float4*>(op + 4) = r1;
    }
}

extern "C" void kernel_launch(void* const* d_in, const int* in_sizes, int n_in,
                              void* d_out, int out_size, void* d_ws, size_t ws_size,
                              hipStream_t stream) {
    const float* src        = (const float*)d_in[0];
    const float* ln_w       = (const float*)d_in[1];
    const float* ln_b       = (const float*)d_in[2];
    const float* in_proj_w  = (const float*)d_in[3];
    const float* in_proj_b  = (const float*)d_in[4];
    const float* out_proj_w = (const float*)d_in[5];
    const float* out_proj_b = (const float*)d_in[6];
    float* out = (float*)d_out;

    unsigned char* ws = (unsigned char*)d_ws;
    unsigned short* x2    = (unsigned short*)(ws + 0);          //  8,388,608 B
    unsigned short* wIn   = (unsigned short*)(ws + 8388608);    //    393,216 B
    unsigned short* wOut  = (unsigned short*)(ws + 8781824);    //    131,072 B
    float*          qkv   = (float*)        (ws + 8912896);     // 50,331,648 B
    float*          padkv = (float*)        (ws + 59244544);    //      2,048 B
    unsigned short* obf   = (unsigned short*)(ws + 59246592);   //  8,388,608 B

    k_convert_bf16<<<768, 256, 0, stream>>>(in_proj_w, wIn, 768 * 256);
    k_convert_bf16<<<256, 256, 0, stream>>>(out_proj_w, wOut, 256 * 256);
    k_pad_kv<<<2, 256, 0, stream>>>(ln_b, in_proj_w, in_proj_b, padkv);
    k_layernorm<<<2048, 256, 0, stream>>>(src, ln_w, ln_b, x2);           // 16384 waves
    k_gemm_qkv<<<1536, 256, 0, stream>>>(x2, wIn, in_proj_b, qkv);        // 12288 waves
    k_attention<<<2048, 256, 0, stream>>>(qkv, padkv, obf);               // 16384 waves
    k_gemm_out<<<512, 256, 0, stream>>>(obf, wOut, out_proj_b, src, out); //  4096 waves
}
